// SimpleModel_73100343378001
// MI455X (gfx1250) — matrix operand order repaired and checked by reading, not compiled, not run on hardware
//
#include <hip/hip_runtime.h>

// ---------------- problem constants ----------------
#define B_  4
#define T_  1024
#define D_  1024
#define H_  16
#define HD_ 64
#define V_  32000

// ---------------- WMMA fragment types (CDNA5, wave32) ----------------
typedef __attribute__((ext_vector_type(16))) __bf16 v16bf;
typedef __attribute__((ext_vector_type(8)))  __bf16 v8bf;
typedef __attribute__((ext_vector_type(8)))  float  v8f;
typedef __attribute__((ext_vector_type(4)))  unsigned uint32x4;
typedef __attribute__((ext_vector_type(8)))  int      int32x8;
typedef __attribute__((ext_vector_type(4)))  int      int32x4;

union V16U { v16bf v; v8bf h[2]; };

__device__ __forceinline__ __bf16 f2bf(float f) {
  union { float f; unsigned u; } a; a.f = f;
  unsigned u = a.u;
  u += 0x7FFFu + ((u >> 16) & 1u);           // round-to-nearest-even
  union { unsigned short s; __bf16 b; } r;
  r.s = (unsigned short)(u >> 16);
  return r.b;
}

// A fragment: 16x32 bf16, row-major source. ISA layout: lane m=l&15, half=l>>4,
// elems 0..7 -> K = 8*half + 0..7 ; elems 8..15 -> K = 16 + 8*half + 0..7.
__device__ __forceinline__ v16bf load_a_frag(const __bf16* base /*A + row0*lda + k0*/,
                                             int lda, int lane) {
  const int m = lane & 15, hlf = lane >> 4;
  const __bf16* p = base + (size_t)m * lda + 8 * hlf;
  V16U u;
  u.h[0] = *(const v8bf*)(p);
  u.h[1] = *(const v8bf*)(p + 16);
  return u.v;
}

// B fragment: 32x16 bf16 from B^T stored row-major [N,K]. ISA layout: lane n=l&15,
// half=l>>4, elems 0..7 -> K = 8*half + 0..7 ; elems 8..15 -> K = 16 + 8*half + 0..7.
__device__ __forceinline__ v16bf load_b_frag(const __bf16* base /*Bt + n0*ldb + k0*/,
                                             int ldb, int lane) {
  const int n = lane & 15, hlf = lane >> 4;
  const __bf16* p = base + (size_t)n * ldb + 8 * hlf;
  V16U u;
  u.h[0] = *(const v8bf*)(p);
  u.h[1] = *(const v8bf*)(p + 16);
  return u.v;
}

__device__ __forceinline__ v8f wmma_bf16(v16bf a, v16bf b, v8f c) {
  return __builtin_amdgcn_wmma_f32_16x16x32_bf16(false, a, false, b, (short)0, c,
                                                 false, false);
}

// ---------------- prep kernels ----------------
__global__ void embed_kernel(const int* __restrict__ idx,
                             const float* __restrict__ tok,
                             const float* __restrict__ pos,
                             __bf16* __restrict__ X) {
  const int row = blockIdx.x;               // b*T + t
  const int tok_id = idx[row];
  const int t = row & (T_ - 1);
  const float* tb = tok + (size_t)tok_id * D_;
  const float* pb = pos + (size_t)t * D_;
  for (int d = threadIdx.x; d < D_; d += 256)
    X[(size_t)row * D_ + d] = f2bf(tb[d] + pb[d]);
}

// Wq [H,D,HD] fp32 -> Wt [Dout, Din] bf16 with Dout col j = h*64+e (B^T layout)
__global__ void repack_qkv_kernel(const float* __restrict__ W, __bf16* __restrict__ Wt) {
  const int j = blockIdx.x;
  const int h = j >> 6, e = j & 63;
  const float* src = W + (size_t)h * D_ * HD_ + e;
  for (int d = threadIdx.x; d < D_; d += 256)
    Wt[(size_t)j * D_ + d] = f2bf(src[(size_t)d * HD_]);
}

// W [K,N] fp32 -> Wt [N,K] bf16 (transpose)
__global__ void repack_t_kernel(const float* __restrict__ W, __bf16* __restrict__ Wt,
                                int K, int N) {
  const int n = blockIdx.x;
  for (int k = threadIdx.x; k < K; k += 256)
    Wt[(size_t)n * K + k] = f2bf(W[(size_t)k * N + n]);
}

// ---------------- generic WMMA GEMM: C = act(A @ B^T + bias) ----------------
__global__ __launch_bounds__(256)
void gemm_bf16_kernel(const __bf16* __restrict__ A, const __bf16* __restrict__ Bt,
                      const float* __restrict__ bias,
                      float* __restrict__ Cf, __bf16* __restrict__ Cb,
                      int M, int N, int K, int relu) {
  const int lane = threadIdx.x & 31;
  const int wave = threadIdx.x >> 5;        // 0..7
  const int wN = wave & 3, wM = wave >> 2;  // 2 (M) x 4 (N) waves
  const int n0 = blockIdx.x * 128 + wN * 32;
  const int m0 = blockIdx.y * 128 + wM * 64;
  if (m0 + 64 > M || n0 + 32 > N) return;

  v8f acc[4][2];
#pragma unroll
  for (int mt = 0; mt < 4; ++mt)
#pragma unroll
    for (int nt = 0; nt < 2; ++nt)
#pragma unroll
      for (int r = 0; r < 8; ++r) acc[mt][nt][r] = 0.0f;

  for (int k0 = 0; k0 < K; k0 += 32) {
    // unconditional speculative stream-ahead (global_prefetch_b8, TH=0: OOB dropped)
    __builtin_prefetch(A + (size_t)(m0 + (lane & 15)) * K + k0 + 32, 0, 1);
    __builtin_prefetch(Bt + (size_t)(n0 + (lane & 15)) * K + k0 + 32, 0, 1);

    v16bf afr[4];
#pragma unroll
    for (int mt = 0; mt < 4; ++mt)
      afr[mt] = load_a_frag(A + (size_t)(m0 + 16 * mt) * K + k0, K, lane);
    v16bf bfr[2];
#pragma unroll
    for (int nt = 0; nt < 2; ++nt)
      bfr[nt] = load_b_frag(Bt + (size_t)(n0 + 16 * nt) * K + k0, K, lane);
#pragma unroll
    for (int mt = 0; mt < 4; ++mt)
#pragma unroll
      for (int nt = 0; nt < 2; ++nt)
        acc[mt][nt] = wmma_bf16(afr[mt], bfr[nt], acc[mt][nt]);
  }

  const int n16 = lane & 15, hlf = lane >> 4;
#pragma unroll
  for (int nt = 0; nt < 2; ++nt) {
    const int col = n0 + 16 * nt + n16;
    const float bv = bias ? bias[col] : 0.0f;
#pragma unroll
    for (int mt = 0; mt < 4; ++mt) {
#pragma unroll
      for (int r = 0; r < 8; ++r) {
        float v = acc[mt][nt][r] + bv;
        if (relu) v = fmaxf(v, 0.0f);
        const size_t row = (size_t)(m0 + 16 * mt + r + 8 * hlf);
        if (Cf) Cf[row * N + col] = v;
        if (Cb) Cb[row * N + col] = f2bf(v);
      }
    }
  }
}

// ---------------- flash attention ----------------
// One workgroup = 4 waves = 4 consecutive 16-row query blocks of the SAME (b,h).
// The 32-key V tile is shared: wave 0 stages it via the Tensor Data Mover
// (tensor_load_to_lds + s_wait_tensorcnt), then the block transposes it in LDS.
__global__ __launch_bounds__(128)
void attention_kernel(const __bf16* __restrict__ Q, const __bf16* __restrict__ Km,
                      const __bf16* __restrict__ Vm, __bf16* __restrict__ O) {
  __shared__ __align__(64) __bf16 sVs[32 * 64];   // TDM staging, row-major [s][e]
  __shared__ __align__(64) __bf16 sVt[64 * 32];   // transposed      [e][s]
  __shared__ __align__(64) __bf16 sP[4][16 * 32]; // per-wave P tile [t][s]

  const int lane = threadIdx.x & 31;
  const int w = threadIdx.x >> 5;
  const int qgrp = blockIdx.x & 15;          // group of 4 query blocks
  const int bh = blockIdx.x >> 4;            // (b,h) instance
  const int hh = bh % H_;
  const int bb = bh / H_;
  const int t0 = (qgrp * 4 + w) * 16;        // this wave's query block
  const int t0_max = qgrp * 64 + 48;         // wave 3's query block (loop bound)
  const int n16 = lane & 15, hlf = lane >> 4;

  const __bf16* Qb = Q + (size_t)bb * T_ * D_ + (size_t)hh * HD_;
  const __bf16* Kb = Km + (size_t)bb * T_ * D_ + (size_t)hh * HD_;
  const __bf16* Vb = Vm + (size_t)bb * T_ * D_ + (size_t)hh * HD_;
  __bf16* Pl = sP[w];

  v16bf aq[2];
#pragma unroll
  for (int kt = 0; kt < 2; ++kt)
    aq[kt] = load_a_frag(Qb + (size_t)t0 * D_ + kt * 32, D_, lane);

  v8f oacc[4];
  float mrow[8], lsum[8];
#pragma unroll
  for (int nt = 0; nt < 4; ++nt)
#pragma unroll
    for (int r = 0; r < 8; ++r) oacc[nt][r] = 0.0f;
#pragma unroll
  for (int r = 0; r < 8; ++r) { mrow[r] = -1e30f; lsum[r] = 0.0f; }

  for (int s0 = 0; s0 <= t0_max + 15; s0 += 32) {
    __syncthreads();  // previous iteration's reads of sVs/sVt complete

    // ---- stage V[s0..s0+31, 0..63] into LDS via the Tensor Data Mover
    if (w == 0) {
#if __has_builtin(__builtin_amdgcn_tensor_load_to_lds)
      const unsigned long long gaddr =
          (unsigned long long)(uintptr_t)(const void*)(Vb + (size_t)s0 * D_);
      const unsigned ldsoff = (unsigned)(uintptr_t)(void*)sVs;
      uint32x4 g0;                         // D# group 0 (ISA 8.3)
      g0[0] = 1u;                          // count=1, user mode
      g0[1] = ldsoff;                      // lds_addr (bytes)
      g0[2] = (unsigned)gaddr;             // global_addr[31:0]
      g0[3] = (unsigned)((gaddr >> 32) & 0x01FFFFFFu) | (2u << 30);  // [56:32]|type=2
      int32x8 g1;                          // D# group 1 (ISA 8.4)
      g1[0] = (int)(1u << 16);             // workgroup_mask=0, data_size=1 (2B)
      g1[1] = (int)((unsigned)HD_ << 16);  // tensor_dim0 = 64 (elements)
      g1[2] = (int)((unsigned)T_ << 16);   // dim0_hi=0 | tensor_dim1 = 1024
      g1[3] = (int)((unsigned)HD_ << 16);  // dim1_hi=0 | tile_dim0 = 64
      g1[4] = 32;                          // tile_dim1 = 32, tile_dim2 = 0
      g1[5] = D_;                          // tensor_dim0_stride = 1024 (elements)
      g1[6] = 0;                           // stride0_hi | stride1_lo
      g1[7] = 0;                           // stride1_hi
      int32x4 gz;
      gz[0] = 0; gz[1] = 0; gz[2] = 0; gz[3] = 0;
      int32x8 gz8;                         // extra group (unused for 2D tensors)
      gz8[0] = 0; gz8[1] = 0; gz8[2] = 0; gz8[3] = 0;
      gz8[4] = 0; gz8[5] = 0; gz8[6] = 0; gz8[7] = 0;
      // 6-arg variant (clang-23 / therock-10.0 toolchain)
      __builtin_amdgcn_tensor_load_to_lds(g0, g1, gz, gz, gz8, 0);
      __builtin_amdgcn_s_wait_tensorcnt(0);
#else
      // fallback: cooperative direct global->LDS copy (wave 0 only)
      const __bf16* vrow = Vb + (size_t)(s0 + lane) * D_;
#pragma unroll
      for (int e8 = 0; e8 < 8; ++e8)
        *(v8bf*)(sVs + lane * 64 + e8 * 8) = *(const v8bf*)(vrow + e8 * 8);
#endif
    }
    __syncthreads();  // V tile visible to all waves

    // ---- cooperative transpose sVs[s][e] -> sVt[e][s] (128 threads, 16 elems each)
    {
      const int s = threadIdx.x & 31;
      const int ebase = (threadIdx.x >> 5) * 16;
#pragma unroll
      for (int q = 0; q < 16; ++q)
        sVt[(ebase + q) * 32 + s] = sVs[s * 64 + ebase + q];
    }
    __syncthreads();

    if (s0 <= t0 + 15) {  // wave-uniform guard; no barriers inside
      // ---- scores S = Q K^T / sqrt(HD)
      v8f sc[2];
#pragma unroll
      for (int nt = 0; nt < 2; ++nt) {
#pragma unroll
        for (int r = 0; r < 8; ++r) sc[nt][r] = 0.0f;
#pragma unroll
        for (int kt = 0; kt < 2; ++kt) {
          v16bf bk = load_b_frag(Kb + (size_t)(s0 + 16 * nt) * D_ + kt * 32, D_, lane);
          sc[nt] = wmma_bf16(aq[kt], bk, sc[nt]);
        }
      }

      // ---- scale + causal mask
#pragma unroll
      for (int nt = 0; nt < 2; ++nt)
#pragma unroll
        for (int r = 0; r < 8; ++r) {
          const int scol = s0 + 16 * nt + n16;
          const int trow = t0 + r + 8 * hlf;
          const float v = sc[nt][r] * 0.125f;
          sc[nt][r] = (scol <= trow) ? v : -1e30f;
        }

      // ---- online softmax
      float corr[8];
#pragma unroll
      for (int r = 0; r < 8; ++r) {
        float mx = fmaxf(sc[0][r], sc[1][r]);
#pragma unroll
        for (int off = 1; off <= 8; off <<= 1) mx = fmaxf(mx, __shfl_xor(mx, off, 32));
        const float mnew = fmaxf(mrow[r], mx);
        corr[r] = __expf(mrow[r] - mnew);
        mrow[r] = mnew;
        const float p0 = __expf(sc[0][r] - mnew);
        const float p1 = __expf(sc[1][r] - mnew);
        sc[0][r] = p0; sc[1][r] = p1;
        float rs = p0 + p1;
#pragma unroll
        for (int off = 1; off <= 8; off <<= 1) rs += __shfl_xor(rs, off, 32);
        lsum[r] = lsum[r] * corr[r] + rs;
      }
#pragma unroll
      for (int nt = 0; nt < 4; ++nt)
#pragma unroll
        for (int r = 0; r < 8; ++r) oacc[nt][r] *= corr[r];

      // ---- relayout P: C/D layout -> A layout via per-wave LDS (DS in-order)
#pragma unroll
      for (int nt = 0; nt < 2; ++nt)
#pragma unroll
        for (int r = 0; r < 8; ++r)
          Pl[(r + 8 * hlf) * 32 + 16 * nt + n16] = f2bf(sc[nt][r]);

      V16U ap;
      ap.h[0] = *(const v8bf*)(Pl + n16 * 32 + 8 * hlf);
      ap.h[1] = *(const v8bf*)(Pl + n16 * 32 + 16 + 8 * hlf);

      // ---- O += P @ V  (one k=32 WMMA per 16-col tile)
#pragma unroll
      for (int nt = 0; nt < 4; ++nt) {
        v16bf bv = load_b_frag(sVt + (size_t)(16 * nt) * 32, 32, lane);
        oacc[nt] = wmma_bf16(ap.v, bv, oacc[nt]);
      }
    }
  }

  // ---- normalize and write out [b, t, h*64+e]
  __bf16* Ob = O + (size_t)(bb * T_ + t0) * D_ + (size_t)hh * HD_;
#pragma unroll
  for (int r = 0; r < 8; ++r) {
    const float inv = 1.0f / lsum[r];
#pragma unroll
    for (int nt = 0; nt < 4; ++nt)
      Ob[(size_t)(r + 8 * hlf) * D_ + 16 * nt + n16] = f2bf(oacc[nt][r] * inv);
  }
}

// ---------------- host launch ----------------
extern "C" void kernel_launch(void* const* d_in, const int* in_sizes, int n_in,
                              void* d_out, int out_size, void* d_ws, size_t ws_size,
                              hipStream_t stream) {
  const int*   idx  = (const int*)d_in[0];
  const float* tok  = (const float*)d_in[1];
  const float* pos  = (const float*)d_in[2];
  const float* Wq   = (const float*)d_in[3];
  const float* Wk   = (const float*)d_in[4];
  const float* Wv   = (const float*)d_in[5];
  const float* W1   = (const float*)d_in[6];
  const float* b1   = (const float*)d_in[7];
  const float* W2   = (const float*)d_in[8];
  const float* b2   = (const float*)d_in[9];
  const float* Wout = (const float*)d_in[10];
  const float* bout = (const float*)d_in[11];
  float* logits = (float*)d_out;

  char* w = (char*)d_ws;
  size_t off = 0;
  auto alloc = [&](size_t bytes) -> void* {
    void* p = w + off;
    off = (off + bytes + 255) & ~(size_t)255;
    return p;
  };
  const size_t szDD = (size_t)D_ * D_ * 2;
  const size_t szX  = (size_t)B_ * T_ * D_ * 2;
  __bf16* WqT   = (__bf16*)alloc(szDD);
  __bf16* WkT   = (__bf16*)alloc(szDD);
  __bf16* WvT   = (__bf16*)alloc(szDD);
  __bf16* W1T   = (__bf16*)alloc(szDD);
  __bf16* W2T   = (__bf16*)alloc(szDD);
  __bf16* WoutT = (__bf16*)alloc((size_t)V_ * D_ * 2);  // 65.5 MB, L2-resident
  __bf16* Xb    = (__bf16*)alloc(szX);
  __bf16* Qb    = (__bf16*)alloc(szX);
  __bf16* Kb    = (__bf16*)alloc(szX);
  __bf16* Vbuf  = (__bf16*)alloc(szX);
  __bf16* Att   = (__bf16*)alloc(szX);
  __bf16* H1    = Xb;   // x dead after QKV
  __bf16* H2    = Qb;   // Q dead after attention
  if (off > ws_size) return;

  const int M = B_ * T_;  // 4096

  embed_kernel<<<M, 256, 0, stream>>>(idx, tok, pos, Xb);

  repack_qkv_kernel<<<D_, 256, 0, stream>>>(Wq, WqT);
  repack_qkv_kernel<<<D_, 256, 0, stream>>>(Wk, WkT);
  repack_qkv_kernel<<<D_, 256, 0, stream>>>(Wv, WvT);
  repack_t_kernel<<<D_, 256, 0, stream>>>(W1, W1T, D_, D_);
  repack_t_kernel<<<D_, 256, 0, stream>>>(W2, W2T, D_, D_);
  repack_t_kernel<<<V_, 256, 0, stream>>>(Wout, WoutT, D_, V_);

  dim3 gD(D_ / 128, M / 128);
  gemm_bf16_kernel<<<gD, 256, 0, stream>>>(Xb, WqT, nullptr, nullptr, Qb, M, D_, D_, 0);
  gemm_bf16_kernel<<<gD, 256, 0, stream>>>(Xb, WkT, nullptr, nullptr, Kb, M, D_, D_, 0);
  gemm_bf16_kernel<<<gD, 256, 0, stream>>>(Xb, WvT, nullptr, nullptr, Vbuf, M, D_, D_, 0);

  attention_kernel<<<(B_ * H_ * (T_ / 16)) / 4, 128, 0, stream>>>(Qb, Kb, Vbuf, Att);

  gemm_bf16_kernel<<<gD, 256, 0, stream>>>(Att, W1T, b1, nullptr, H1, M, D_, D_, 1);
  gemm_bf16_kernel<<<gD, 256, 0, stream>>>(H1, W2T, b2, nullptr, H2, M, D_, D_, 1);

  dim3 gV(V_ / 128, M / 128);
  gemm_bf16_kernel<<<gV, 256, 0, stream>>>(H2, WoutT, bout, logits, nullptr, M, V_, D_, 0);
}
